// Conv_7164005449944
// MI455X (gfx1250) — compile-verified, matrix operand order checked by reading, not run on hardware
//
#include <hip/hip_runtime.h>
#include <hip/hip_bf16.h>

typedef __attribute__((ext_vector_type(16))) __bf16 bf16x16;
typedef __attribute__((ext_vector_type(8)))  __bf16 bf16x8;
typedef __attribute__((ext_vector_type(8)))  float  f32x8;

#define INV640 0.039528470752104741f   // 1/sqrt(640)
#define ISQ3   0.57735026918962584f
#define ISQ2   0.70710678118654757f

// ---------------------------------------------------------------------------
// Pack the 6 FCTP weight tensors (64x10x64 f32, index [u][v][w]) into bf16
// WMMA B-fragment order with K reordered as k = v*64 + u, plus tp_weight
// (32x256) into B-fragment order (K=32 exactly one chunk).
// Fragment (kc, nc): lane l (<16) holds col n = nc*16 + l, K-offsets
// {0..7, 16..23}; lane l>=16 holds K-offsets {8..15, 24..31}.
// ---------------------------------------------------------------------------
__global__ void pack_kernel(const float* __restrict__ W0, const float* __restrict__ W1,
                            const float* __restrict__ W2, const float* __restrict__ W3,
                            const float* __restrict__ W4, const float* __restrict__ W5,
                            const float* __restrict__ tp,
                            __bf16* __restrict__ wp, __bf16* __restrict__ tp_pk) {
  int t = blockIdx.x * blockDim.x + threadIdx.x;
  if (t < 6 * 40960) {
    int widx = t / 40960, rem = t - widx * 40960;
    const float* W = widx == 0 ? W0 : widx == 1 ? W1 : widx == 2 ? W2
                   : widx == 3 ? W3 : widx == 4 ? W4 : W5;
    int frag = rem >> 9;          // 512 bf16 per fragment
    int within = rem & 511;
    int kc = frag >> 2, nc = frag & 3;
    int lane = within >> 4, j = within & 15;
    int h = lane >> 4, nn = lane & 15;
    int koff = (j < 8) ? (8 * h + j) : (16 + 8 * h + (j - 8));
    int v = kc >> 1;
    int u = ((kc & 1) << 5) + koff;
    wp[t] = (__bf16)W[(u * 10 + v) * 64 + nc * 16 + nn];
  } else {
    int t2 = t - 6 * 40960;
    if (t2 < 8192) {
      int nc = t2 >> 9, within = t2 & 511;
      int lane = within >> 4, j = within & 15;
      int h = lane >> 4, nn = lane & 15;
      int k = (j < 8) ? (8 * h + j) : (16 + 8 * h + (j - 8));
      tp_pk[t2] = (__bf16)tp[k * 256 + nc * 16 + nn];
    }
  }
}

__global__ void zero_kernel(float4* __restrict__ p, int n4) {
  int i = blockIdx.x * blockDim.x + threadIdx.x;
  if (i < n4) p[i] = float4{0.f, 0.f, 0.f, 0.f};
}

// ---------------------------------------------------------------------------
// FCTP: out[n, :] = concat(o0, o1) with
//   o0[n,w]   = scale * sum_{u,v} x0[n,u]  z[n,v] W0[u,v,w]
//   o1[n,w,k] = scale * sum_{u,v} x1[n,u,k] z[n,v] W1[u,v,w]
// One wave per 32-node M-tile (two 16-row A fragments share each B fragment).
// K = 640 = 20 chunks of 32 with k = v*64 + u: v constant within a chunk ->
// A fragment = contiguous bf16 x-run * z broadcast (v_pk_mul_bf16).
// Per chunk: 8 B-loads, 16 pk_mul, 8 WMMAs.  640 WMMAs / 32-node tile.
// ---------------------------------------------------------------------------
template <int ACC>
__global__ void __launch_bounds__(128) fctp_kernel(
    const float* __restrict__ X, const float* __restrict__ Z,
    const __bf16* __restrict__ Wp0, const __bf16* __restrict__ Wp1,
    float* __restrict__ out, float scale, int nNodes) {
  __shared__ __align__(16) __bf16 xsc[4][4][32][64];  // [wave][chan][row][u]
  __shared__ __align__(16) __bf16 zb[4][32][16];      // [wave][row][v] (padded)
  const int wave = threadIdx.x >> 5;
  const int lane = threadIdx.x & 31;
  const int n_base = (blockIdx.x * 4 + wave) * 32;
  if (n_base >= nNodes) return;   // no cross-wave LDS sharing, safe

  // Stage tile: x -> bf16 channel-major, z -> bf16
  for (int idx = lane; idx < 32 * 256; idx += 32) {
    int r = idx >> 8, f = idx & 255;
    int n = n_base + r;
    float val = (n < nNodes) ? X[(size_t)n * 256 + f] : 0.f;
    int c, u;
    if (f < 64) { c = 0; u = f; }
    else { int g = f - 64; u = g / 3; c = 1 + (g - u * 3); }
    xsc[wave][c][r][u] = (__bf16)val;
  }
  for (int idx = lane; idx < 512; idx += 32) {
    int r = idx >> 4, v = idx & 15;
    int n = n_base + r;
    zb[wave][r][v] = (__bf16)((v < 10 && n < nNodes) ? Z[n * 10 + v] : 0.f);
  }

  const int r = lane & 15;
  const int h = lane >> 4;
  const bool full = (n_base + 32 <= nNodes);

  for (int c = 0; c < 4; ++c) {
    const __bf16* Wp = (c == 0) ? Wp0 : Wp1;
    f32x8 acc[2][4] = {{f32x8{}, f32x8{}, f32x8{}, f32x8{}},
                       {f32x8{}, f32x8{}, f32x8{}, f32x8{}}};
    for (int kc = 0; kc < 20; ++kc) {
      const int v  = kc >> 1;
      const int u0 = (kc & 1) << 5;
      // Two 16-row A fragments (rows r and r+16 of the 32-node tile)
      bf16x16 a[2];
#pragma unroll
      for (int mh = 0; mh < 2; ++mh) {
        int row = r + 16 * mh;
        float zf = (float)zb[wave][row][v];
        bf16x8 r1 = *(const bf16x8*)&xsc[wave][c][row][u0 + 8 * h];
        bf16x8 r2 = *(const bf16x8*)&xsc[wave][c][row][u0 + 16 + 8 * h];
#pragma unroll
        for (int j = 0; j < 8; ++j) {
          a[mh][j]     = (__bf16)((float)r1[j] * zf);
          a[mh][j + 8] = (__bf16)((float)r2[j] * zf);
        }
      }
      const __bf16* wb = Wp + ((size_t)(kc * 4) * 32 + lane) * 16;
#pragma unroll
      for (int nc = 0; nc < 4; ++nc) {
        bf16x16 b = *(const bf16x16*)(wb + nc * 512);
        acc[0][nc] = __builtin_amdgcn_wmma_f32_16x16x32_bf16(
            false, a[0], false, b, (short)0, acc[0][nc], false, false);
        acc[1][nc] = __builtin_amdgcn_wmma_f32_16x16x32_bf16(
            false, a[1], false, b, (short)0, acc[1][nc], false, false);
      }
    }
    // C/D layout: VGPR r8, lane l -> row M = r8 + 8*(l>>4), col N = l&15
#pragma unroll
    for (int mh = 0; mh < 2; ++mh) {
#pragma unroll
      for (int nc = 0; nc < 4; ++nc) {
        int wcol = nc * 16 + r;
        int feat = (c == 0) ? wcol : (64 + 3 * wcol + (c - 1));
#pragma unroll
        for (int r8 = 0; r8 < 8; ++r8) {
          int n = n_base + 16 * mh + r8 + 8 * h;
          float val = acc[mh][nc][r8] * scale;
          float* p = out + (size_t)n * 256 + feat;
          if (full) {
            if (ACC) *p += val; else *p = val;
          } else if (n < nNodes) {
            if (ACC) *p += val; else *p = val;
          }
        }
      }
    }
  }
}

// ---------------------------------------------------------------------------
// Edge kernel: per wave, one 16-edge tile.
// Phase 1: w(16x256) = len_emb(16x32) @ tp(32x256) via 16 WMMAs -> LDS.
// Phase 2: gather xl[src], equivariant products, 4 atomic adds per (e,u).
// ---------------------------------------------------------------------------
__global__ void __launch_bounds__(64) edge_kernel(
    const float* __restrict__ xl, const int* __restrict__ esrc,
    const int* __restrict__ edst, const float* __restrict__ lemb,
    const float* __restrict__ eattr, const __bf16* __restrict__ tp_pk,
    float* __restrict__ agg, int nEdges) {
  __shared__ __align__(16) float wlds[2][16][256];
  const int wave = threadIdx.x >> 5;
  const int lane = threadIdx.x & 31;
  const int e_base = (blockIdx.x * 2 + wave) * 16;
  if (e_base >= nEdges) return;

  const int r = lane & 15;
  const int h = lane >> 4;

  // Phase 1: A fragment from edge_len_emb (row r of tile), K = 32 (one chunk)
  bf16x16 a;
  {
    int e = e_base + r;
    if (e < nEdges) {
      const float* row = lemb + (size_t)e * 32;
#pragma unroll
      for (int j = 0; j < 8; ++j) {
        a[j]     = (__bf16)row[8 * h + j];
        a[j + 8] = (__bf16)row[16 + 8 * h + j];
      }
    } else {
#pragma unroll
      for (int j = 0; j < 16; ++j) a[j] = (__bf16)0.f;
    }
  }
#pragma unroll
  for (int nc = 0; nc < 16; ++nc) {
    bf16x16 b = *(const bf16x16*)(tp_pk + ((size_t)nc * 32 + lane) * 16);
    f32x8 d = {};
    d = __builtin_amdgcn_wmma_f32_16x16x32_bf16(
        false, a, false, b, (short)0, d, false, false);
#pragma unroll
    for (int r8 = 0; r8 < 8; ++r8)
      wlds[wave][r8 + 8 * h][nc * 16 + r] = d[r8];
  }
  // same-wave DS ordering: stores complete before loads below (compiler waits)

  // Phase 2: 16 edges x 64 features = 1024 work items / 32 lanes
  for (int i = 0; i < 32; ++i) {
    int el = i >> 1;
    int u  = ((i & 1) << 5) | lane;
    int e  = e_base + el;
    if (e >= nEdges) break;   // uniform across wave
    int src = esrc[e], dst = edst[e];
    const float4 sh = *(const float4*)(eattr + (size_t)e * 4);
    float w1 = wlds[wave][el][u];
    float w2 = wlds[wave][el][64 + u];
    float w3 = wlds[wave][el][128 + u];
    float w4 = wlds[wave][el][192 + u];
    const float* xr = xl + (size_t)src * 256;
    float x0 = xr[u];
    float xa = xr[64 + 3 * u];
    float xb = xr[64 + 3 * u + 1];
    float xc = xr[64 + 3 * u + 2];
    float dot = xa * sh.y + xb * sh.z + xc * sh.w;
    float e0  = ISQ2 * (w1 * x0 * sh.x + ISQ3 * w4 * dot);
    float e1x = ISQ2 * (w2 * x0 * sh.y + w3 * xa * sh.x);
    float e1y = ISQ2 * (w2 * x0 * sh.z + w3 * xb * sh.x);
    float e1z = ISQ2 * (w2 * x0 * sh.w + w3 * xc * sh.x);
    float* ap = agg + (size_t)dst * 256;
    unsafeAtomicAdd(ap + u, e0);
    unsafeAtomicAdd(ap + 64 + 3 * u, e1x);
    unsafeAtomicAdd(ap + 64 + 3 * u + 1, e1y);
    unsafeAtomicAdd(ap + 64 + 3 * u + 2, e1z);
  }
}

extern "C" void kernel_launch(void* const* d_in, const int* in_sizes, int n_in,
                              void* d_out, int out_size, void* d_ws, size_t ws_size,
                              hipStream_t stream) {
  (void)n_in; (void)out_size; (void)ws_size;
  const float* x    = (const float*)d_in[0];
  const float* z    = (const float*)d_in[1];
  const int*   esrc = (const int*)d_in[2];
  const int*   edst = (const int*)d_in[3];
  const float* lemb = (const float*)d_in[4];
  const float* eatt = (const float*)d_in[5];
  const float* Wsi0 = (const float*)d_in[6];
  const float* Wsi1 = (const float*)d_in[7];
  const float* Wl10 = (const float*)d_in[8];
  const float* Wl11 = (const float*)d_in[9];
  const float* Wl20 = (const float*)d_in[10];
  const float* Wl21 = (const float*)d_in[11];
  const float* tpw  = (const float*)d_in[12];
  float* out = (float*)d_out;

  const int nNodes = in_sizes[0] / 256;   // 50000
  const int nEdges = in_sizes[2];         // 500000

  char* ws = (char*)d_ws;
  __bf16* wp    = (__bf16*)ws;                        // 6 * 40960 bf16 = 480 KB
  __bf16* tp_pk = wp + 6 * 40960;                     // 8192 bf16 = 16 KB
  float*  xl    = (float*)(ws + (1u << 20));          // nNodes*256 f32
  float*  agg   = (float*)(ws + (1u << 20) + (size_t)nNodes * 256 * 4);

  const int nFeat = nNodes * 256;
  const int ntiles = (nNodes + 31) / 32;
  const int fctp_blocks = (ntiles + 3) / 4;
  const int etiles = (nEdges + 15) / 16;
  const int edge_blocks = (etiles + 1) / 2;

  pack_kernel<<<(6 * 40960 + 8192 + 255) / 256, 256, 0, stream>>>(
      Wsi0, Wsi1, Wl10, Wl11, Wl20, Wl21, tpw, wp, tp_pk);
  zero_kernel<<<(nFeat / 4 + 255) / 256, 256, 0, stream>>>((float4*)agg, nFeat / 4);
  fctp_kernel<0><<<fctp_blocks, 128, 0, stream>>>(x, z, wp + 0 * 40960, wp + 1 * 40960,
                                                  out, INV640, nNodes);
  fctp_kernel<0><<<fctp_blocks, 128, 0, stream>>>(x, z, wp + 2 * 40960, wp + 3 * 40960,
                                                  xl, INV640, nNodes);
  edge_kernel<<<edge_blocks, 64, 0, stream>>>(xl, esrc, edst, lemb, eatt,
                                              tp_pk, agg, nEdges);
  fctp_kernel<1><<<fctp_blocks, 128, 0, stream>>>(agg, z, wp + 4 * 40960, wp + 5 * 40960,
                                                  out, INV640 * 0.1f, nNodes);
}
